// Attention_36481452213061
// MI455X (gfx1250) — matrix operand order repaired and checked by reading, not compiled, not run on hardware
//
#include <hip/hip_runtime.h>

#define HIDDEN   2048
#define HEAD_DIM 128
#define N_HEADS  16
#define WINDOW   512
#define BATCH    2
#define SEQ      2048
#define QKV_OUT  ((N_HEADS * 3) * HEAD_DIM)   // 6144
#define M_ROWS   (BATCH * SEQ)                // 4096

typedef __attribute__((ext_vector_type(16))) __bf16 v16bf;
typedef __attribute__((ext_vector_type(8)))  __bf16 v8bf;
typedef __attribute__((ext_vector_type(8)))  float  v8f;
typedef __attribute__((ext_vector_type(4)))  float  f32x4;

__device__ __forceinline__ __bf16 f2bf(float f) {
  unsigned u = __builtin_bit_cast(unsigned, f);
  unsigned r = (u + 0x7FFFu + ((u >> 16) & 1u)) >> 16;  // round-to-nearest-even
  unsigned short h = (unsigned short)r;
  return __builtin_bit_cast(__bf16, h);
}

__device__ __forceinline__ v8f wmma_bf16(v16bf a, v16bf b, v8f c) {
  // D(16x16,f32) = A(16x32,bf16) * B(32x16,bf16) + C
  return __builtin_amdgcn_wmma_f32_16x16x32_bf16(
      /*neg_a=*/false, a, /*neg_b=*/false, b,
      /*c_mod=*/(short)0, c, /*reuse_a=*/false, /*reuse_b=*/false);
}

__device__ __forceinline__ v8f v8f_zero() {
  v8f z = {0.f, 0.f, 0.f, 0.f, 0.f, 0.f, 0.f, 0.f};
  return z;
}

// A fragment (16x32 bf16): lanes 0-15 hold K [k0..k0+15];
// lanes 16-31 hold K [k0+16..k0+31].
__device__ __forceinline__ v16bf load_a_frag(const __bf16* row, int k0, int hi) {
  return *(const v16bf*)(row + k0 + hi * 16);
}

// B fragment (32x16 bf16): lane = column; lanes 0-15 K [k0..k0+15],
// lanes 16-31 K [k0+16..k0+31]. Column data contiguous in memory.
__device__ __forceinline__ v16bf load_b_frag(const __bf16* col, int k0, int hi) {
  return *(const v16bf*)(col + k0 + hi * 16);
}

// ---------------------------------------------------------------------------
// Bulk f32 -> bf16 conversion (one-shot; removes conversions from GEMM loop)
// ---------------------------------------------------------------------------
__global__ __launch_bounds__(256) void f32_to_bf16(
    const float* __restrict__ src, __bf16* __restrict__ dst, long n) {
  long i = ((long)blockIdx.x * 256 + threadIdx.x) * 8;
  if (i + 8 > n) return;
  const f32x4* p = (const f32x4*)(src + i);
  f32x4 x0 = p[0], x1 = p[1];
  v8bf o;
#pragma unroll
  for (int j = 0; j < 4; ++j) { o[j] = f2bf(x0[j]); o[4 + j] = f2bf(x1[j]); }
  *(v8bf*)(dst + i) = o;
}

// ---------------------------------------------------------------------------
// WMMA GEMM: C[M,N] = A[M,K] * B[N,K]^T   (bf16 operands, K-contiguous,
// f32 output). One wave per 64x64 C tile: 4x4 accumulators, K step 32.
// ---------------------------------------------------------------------------
__global__ __launch_bounds__(32) void wmma_gemm_bf16(
    const __bf16* __restrict__ A, const __bf16* __restrict__ Bm,
    float* __restrict__ C, int M, int N, int K) {
  const int lane = threadIdx.x;
  const int hr = lane & 15;             // lane within half-wave
  const int hi = (lane >> 4) & 1;       // which half-wave
  const int m0 = blockIdx.y * 64;
  const int n0 = blockIdx.x * 64;

  v8f acc[4][4];
#pragma unroll
  for (int i = 0; i < 4; ++i)
#pragma unroll
    for (int j = 0; j < 4; ++j) acc[i][j] = v8f_zero();

  const __bf16* arow[4];
  const __bf16* brow[4];
#pragma unroll
  for (int t = 0; t < 4; ++t) {
    arow[t] = A  + (size_t)(m0 + t * 16 + hr) * K;   // A row for this lane's M
    brow[t] = Bm + (size_t)(n0 + t * 16 + hr) * K;   // B column = weight row
  }

  for (int k0 = 0; k0 < K; k0 += 32) {
    v16bf af[4], bf[4];
#pragma unroll
    for (int t = 0; t < 4; ++t) {
      af[t] = load_a_frag(arow[t], k0, hi);
      bf[t] = load_b_frag(brow[t], k0, hi);
      if (k0 + 32 < K) {  // global_prefetch_b8 for next K slab
        __builtin_prefetch(arow[t] + k0 + 32, 0, 1);
        __builtin_prefetch(brow[t] + k0 + 32, 0, 1);
      }
    }
#pragma unroll
    for (int i = 0; i < 4; ++i)
#pragma unroll
      for (int j = 0; j < 4; ++j) acc[i][j] = wmma_bf16(af[i], bf[j], acc[i][j]);
  }

  // C layout: VGPR r -> row (r + 8*hi), column = hr within tile
#pragma unroll
  for (int i = 0; i < 4; ++i) {
    const int mbase = m0 + i * 16 + hi * 8;
#pragma unroll
    for (int j = 0; j < 4; ++j) {
      const int col = n0 + j * 16 + hr;
#pragma unroll
      for (int r = 0; r < 8; ++r)
        C[(size_t)(mbase + r) * N + col] = acc[i][j][r];
    }
  }
}

// ---------------------------------------------------------------------------
// RoPE + split: qkv f32 [b*S, 48*128] -> Q,K bf16 [b,h,s,d] (1/sqrt(d) folded
// into Q), V bf16 transposed [b,h,d,s] for coalesced PV B-fragments.
// ---------------------------------------------------------------------------
__global__ __launch_bounds__(HEAD_DIM) void rope_split(
    const float* __restrict__ qkv, const float* __restrict__ cosT,
    const float* __restrict__ sinT, __bf16* __restrict__ Q,
    __bf16* __restrict__ Kd, __bf16* __restrict__ Vt) {
  const int row = blockIdx.x;            // b*SEQ + s
  const int b = row / SEQ;
  const int s = row - b * SEQ;
  const int d = threadIdx.x;             // 0..127
  const float c  = cosT[(size_t)s * HEAD_DIM + d];
  const float sn = sinT[(size_t)s * HEAD_DIM + d];
  const float* base = qkv + (size_t)row * QKV_OUT;
  const float scale = 0.08838834764831845f;   // 1/sqrt(128)
  const int   dro = (d < 64) ? (d + 64) : (d - 64);
  const float sgn = (d < 64) ? -1.0f : 1.0f;
#pragma unroll
  for (int h = 0; h < N_HEADS; ++h) {
    const size_t bh = (size_t)(b * N_HEADS + h);
    float q  = base[h * HEAD_DIM + d];
    float qr = sgn * base[h * HEAD_DIM + dro];
    Q[(bh * SEQ + s) * HEAD_DIM + d] = f2bf((q * c + qr * sn) * scale);
    float k  = base[(N_HEADS + h) * HEAD_DIM + d];
    float kr = sgn * base[(N_HEADS + h) * HEAD_DIM + dro];
    Kd[(bh * SEQ + s) * HEAD_DIM + d] = f2bf(k * c + kr * sn);
    float v = base[(2 * N_HEADS + h) * HEAD_DIM + d];
    Vt[(bh * HEAD_DIM + d) * SEQ + s] = f2bf(v);
  }
}

// ---------------------------------------------------------------------------
// Sliding-window attention, flash-style online softmax. One wave per
// (b, h, 16-query tile). Keys in blocks of 32: 8 WMMAs for Q*K^T, mask +
// row max/sum via half-wave shuffles, P restaged through LDS into A-fragment
// layout, 8 WMMAs for P*V. Output stored bf16 [b*S, h*d] for the final GEMM.
// ---------------------------------------------------------------------------
__global__ __launch_bounds__(32) void attn_window(
    const __bf16* __restrict__ Q, const __bf16* __restrict__ Kmat,
    const __bf16* __restrict__ Vt, __bf16* __restrict__ Out) {
  __shared__ __align__(128) __bf16 Plds[16][32];
  const int lane = threadIdx.x;
  const int hr = lane & 15;
  const int hi = (lane >> 4) & 1;
  const int q0 = blockIdx.x * 16;
  const int h  = blockIdx.y;
  const int b  = blockIdx.z;

  const __bf16* Qh = Q    + (size_t)(b * N_HEADS + h) * SEQ * HEAD_DIM;
  const __bf16* Kh = Kmat + (size_t)(b * N_HEADS + h) * SEQ * HEAD_DIM;
  const __bf16* Vh = Vt   + (size_t)(b * N_HEADS + h) * HEAD_DIM * SEQ;

  // Q A-fragments for the whole 16x128 tile (4 fragments of K=32)
  v16bf qf[4];
  const __bf16* qrow = Qh + (size_t)(q0 + hr) * HEAD_DIM;
#pragma unroll
  for (int t = 0; t < 4; ++t) qf[t] = load_a_frag(qrow, t * 32, hi);

  v8f Oacc[8];
#pragma unroll
  for (int t = 0; t < 8; ++t) Oacc[t] = v8f_zero();
  float mrow[8], lrow[8];
#pragma unroll
  for (int r = 0; r < 8; ++r) { mrow[r] = -3.0e38f; lrow[r] = 0.0f; }

  int kmin = q0 - (WINDOW - 1);
  if (kmin < 0) kmin = 0;
  const int kb0  = kmin & ~31;
  const int kend = q0 + 15;

  for (int kb = kb0; kb <= kend; kb += 32) {
    // ---- scores: two 16x16 tiles (keys kb..kb+15 and kb+16..kb+31) ----
    v8f s0 = v8f_zero(), s1 = v8f_zero();
    const __bf16* kcol0 = Kh + (size_t)(kb + hr) * HEAD_DIM;
    const __bf16* kcol1 = Kh + (size_t)(kb + 16 + hr) * HEAD_DIM;
#pragma unroll
    for (int t = 0; t < 4; ++t) {
      v16bf b0 = load_b_frag(kcol0, t * 32, hi);
      v16bf b1 = load_b_frag(kcol1, t * 32, hi);
      s0 = wmma_bf16(qf[t], b0, s0);
      s1 = wmma_bf16(qf[t], b1, s1);
    }

    // ---- mask + online softmax (rows live across one 16-lane half) ----
    float corr[8];
#pragma unroll
    for (int r = 0; r < 8; ++r) {
      const int qr_i = q0 + hi * 8 + r;
      const int d0 = qr_i - (kb + hr);
      const int d1 = qr_i - (kb + 16 + hr);
      const bool ok0 = (d0 >= 0) && (d0 < WINDOW);
      const bool ok1 = (d1 >= 0) && (d1 < WINDOW);
      float v0 = ok0 ? s0[r] : -3.0e38f;
      float v1 = ok1 ? s1[r] : -3.0e38f;
      float rmax = fmaxf(v0, v1);
#pragma unroll
      for (int msk = 8; msk >= 1; msk >>= 1)
        rmax = fmaxf(rmax, __shfl_xor(rmax, msk, 16));
      const float mnew = fmaxf(mrow[r], rmax);
      float p0 = ok0 ? __expf(v0 - mnew) : 0.0f;
      float p1 = ok1 ? __expf(v1 - mnew) : 0.0f;
      float psum = p0 + p1;
#pragma unroll
      for (int msk = 8; msk >= 1; msk >>= 1)
        psum += __shfl_xor(psum, msk, 16);
      const float c = __expf(mrow[r] - mnew);
      lrow[r] = lrow[r] * c + psum;
      mrow[r] = mnew;
      corr[r] = c;
      Plds[hi * 8 + r][hr]      = f2bf(p0);   // C-layout -> LDS
      Plds[hi * 8 + r][16 + hr] = f2bf(p1);
    }
#pragma unroll
    for (int t = 0; t < 8; ++t)
#pragma unroll
      for (int r = 0; r < 8; ++r) Oacc[t][r] = Oacc[t][r] * corr[r];

    __syncthreads();
    // P re-read in A-fragment layout (16x32, K = 32 keys of this block)
    v16bf pf = load_a_frag(&Plds[hr][0], 0, hi);
    // ---- P * V: 8 d-tiles; B column n = Vt row (d), K-contiguous in keys
#pragma unroll
    for (int t = 0; t < 8; ++t) {
      const __bf16* vcol = Vh + (size_t)(t * 16 + hr) * SEQ + kb;
      Oacc[t] = wmma_bf16(pf, load_b_frag(vcol, 0, hi), Oacc[t]);
    }
    __syncthreads();
  }

  // ---- normalize and store bf16 as [b*S, N_HEADS*HEAD_DIM] ----
#pragma unroll
  for (int r = 0; r < 8; ++r) {
    const int qr = q0 + hi * 8 + r;
    const float inv = 1.0f / lrow[r];
#pragma unroll
    for (int t = 0; t < 8; ++t)
      Out[((size_t)b * SEQ + qr) * (size_t)HIDDEN + h * HEAD_DIM + t * 16 + hr] =
          f2bf(Oacc[t][r] * inv);
  }
}

// ---------------------------------------------------------------------------
extern "C" void kernel_launch(void* const* d_in, const int* in_sizes, int n_in,
                              void* d_out, int out_size, void* d_ws, size_t ws_size,
                              hipStream_t stream) {
  const float* hidden = (const float*)d_in[0];
  const float* cosT   = (const float*)d_in[1];
  const float* sinT   = (const float*)d_in[2];
  const float* w_qkv  = (const float*)d_in[3];
  const float* w_o    = (const float*)d_in[4];
  float* out = (float*)d_out;

  char* ws = (char*)d_ws;
  size_t off = 0;
  float* qkv = (float*)(ws + off);                 // [4096,6144] f32
  off += (size_t)M_ROWS * QKV_OUT * sizeof(float);
  __bf16* Hbf = (__bf16*)(ws + off);               // hidden bf16 [4096,2048]
  off += (size_t)M_ROWS * HIDDEN * sizeof(__bf16);
  __bf16* Wqkvbf = (__bf16*)(ws + off);            // [6144,2048]
  off += (size_t)QKV_OUT * HIDDEN * sizeof(__bf16);
  __bf16* Wobf = (__bf16*)(ws + off);              // [2048,2048]
  off += (size_t)HIDDEN * HIDDEN * sizeof(__bf16);
  const size_t bhsd = (size_t)BATCH * N_HEADS * SEQ * HEAD_DIM * sizeof(__bf16);
  __bf16* Qb  = (__bf16*)(ws + off); off += bhsd;
  __bf16* Kb  = (__bf16*)(ws + off); off += bhsd;
  __bf16* Vtb = (__bf16*)(ws + off); off += bhsd;
  __bf16* attn = Hbf;  // Hbf dead after GEMM1; reuse as bf16 attn output

  (void)in_sizes; (void)n_in; (void)out_size; (void)ws_size;

  const long nH = (long)M_ROWS * HIDDEN;
  const long nW = (long)QKV_OUT * HIDDEN;
  const long nO = (long)HIDDEN * HIDDEN;
  // 0) one-shot f32 -> bf16 conversion of GEMM operands
  f32_to_bf16<<<dim3((unsigned)(nH / (8 * 256))), 256, 0, stream>>>(hidden, Hbf, nH);
  f32_to_bf16<<<dim3((unsigned)(nW / (8 * 256))), 256, 0, stream>>>(w_qkv, Wqkvbf, nW);
  f32_to_bf16<<<dim3((unsigned)(nO / (8 * 256))), 256, 0, stream>>>(w_o, Wobf, nO);
  // 1) QKV projection (bf16 x bf16 -> f32)
  wmma_gemm_bf16<<<dim3(QKV_OUT / 64, M_ROWS / 64), 32, 0, stream>>>(
      Hbf, Wqkvbf, qkv, M_ROWS, QKV_OUT, HIDDEN);
  // 2) RoPE + split into bf16 Q, K, V^T
  rope_split<<<dim3(M_ROWS), dim3(HEAD_DIM), 0, stream>>>(
      qkv, cosT, sinT, Qb, Kb, Vtb);
  // 3) sliding-window attention (bf16 out)
  attn_window<<<dim3(SEQ / 16, N_HEADS, BATCH), 32, 0, stream>>>(
      Qb, Kb, Vtb, attn);
  // 4) output projection (f32 out to d_out)
  wmma_gemm_bf16<<<dim3(HIDDEN / 64, M_ROWS / 64), 32, 0, stream>>>(
      attn, Wobf, out, M_ROWS, HIDDEN, HIDDEN);
}
